// Mammoth2DecoderLayer_40467181863530
// MI455X (gfx1250) — matrix operand order repaired and checked by reading, not compiled, not run on hardware
//
#include <hip/hip_runtime.h>
#include <math.h>

// ---------------- types ----------------
typedef __bf16 bf16_t;
typedef __attribute__((ext_vector_type(16))) __bf16 v16bf;
typedef __attribute__((ext_vector_type(8)))  float  v8f;

#define B_   2
#define L_   2048
#define D_   2048
#define H_   16
#define KV_  4
#define HD_  128
#define I_   5504
#define M_   (B_ * L_)        // 4096 token rows

// ---------------- WMMA helpers ----------------
struct FragA { union { v16bf v; uint4 q[2]; }; };

__device__ __forceinline__ v8f vzero() {
  v8f z;
#pragma unroll
  for (int i = 0; i < 8; i++) z[i] = 0.0f;
  return z;
}

// A-operand (16x32 bf16): lane holds row m; K contiguous 16 at (lane>=16 ? 16 : 0),
// caller passes p = row_base + (lane>=16 ? 16 : 0)
__device__ __forceinline__ FragA load_a(const bf16_t* p) {
  FragA f;
  f.q[0] = *(const uint4*)(p);
  f.q[1] = *(const uint4*)(p + 8);
  return f;
}
// B-operand (32x16 bf16): lane holds col n; K contiguous 16 at (lane>=16 ? 16 : 0),
// caller passes p = col_base + (lane>=16 ? 16 : 0)
__device__ __forceinline__ FragA load_b(const bf16_t* p) {
  FragA f;
  f.q[0] = *(const uint4*)(p);
  f.q[1] = *(const uint4*)(p + 8);
  return f;
}

__device__ __forceinline__ v8f wmma_bf16(const FragA& a, const FragA& b, v8f c) {
  return __builtin_amdgcn_wmma_f32_16x16x32_bf16(
      /*neg_a=*/false, a.v, /*neg_b=*/false, b.v,
      /*c_mod=*/(short)0, c, /*reuse_a=*/false, /*reuse_b=*/false);
}

// ---------------- CDNA5 async global->LDS copy (ASYNCcnt path) ----------------
// One instruction per 16B lane-copy; no intermediate VGPR data, no loadcnt stall.
__device__ __forceinline__ void async_copy_b128(unsigned lds_byte_addr,
                                                const void* gaddr) {
  asm volatile("global_load_async_to_lds_b128 %0, %1, off"
               :: "v"(lds_byte_addr), "v"((unsigned long long)(size_t)gaddr)
               : "memory");
}
__device__ __forceinline__ void wait_async0() {
  asm volatile("s_wait_asynccnt 0" ::: "memory");
}

// ---------------- transpose + fp32->bf16 weight conversion ----------------
// in: fp32 [R][C] row-major  -> out: bf16 [C][R] row-major
__global__ __launch_bounds__(256) void transpose_to_bf16_kernel(
    const float* __restrict__ in, bf16_t* __restrict__ out, int R, int C) {
  __shared__ float tile[32][33];
  const int c0 = blockIdx.x * 32;
  const int r0 = blockIdx.y * 32;
  const int tx = threadIdx.x;   // 0..31
  const int ty = threadIdx.y;   // 0..7
#pragma unroll
  for (int i = 0; i < 32; i += 8)
    tile[ty + i][tx] = in[(size_t)(r0 + ty + i) * C + c0 + tx];
  __syncthreads();
#pragma unroll
  for (int i = 0; i < 32; i += 8)
    out[(size_t)(c0 + ty + i) * R + r0 + tx] = (bf16_t)tile[tx][ty + i];
}

// ---------------- fused residual add + RMSNorm ----------------
__global__ __launch_bounds__(256) void add_rmsnorm_kernel(
    const float* __restrict__ x, const float* __restrict__ r,
    const float* __restrict__ w, float* __restrict__ res_out,
    bf16_t* __restrict__ h_out) {
  __shared__ float red[256];
  const int row = blockIdx.x;
  const int tid = threadIdx.x;
  const float* xr = x + (size_t)row * D_;
  const float* rr = r + (size_t)row * D_;
  float v[8];
  float ss = 0.f;
#pragma unroll
  for (int i = 0; i < 8; i++) {
    int idx = tid + i * 256;
    float t = xr[idx] + rr[idx];
    v[i] = t;
    ss += t * t;
  }
  red[tid] = ss;
  __syncthreads();
  for (int s = 128; s > 0; s >>= 1) {
    if (tid < s) red[tid] += red[tid + s];
    __syncthreads();
  }
  const float inv = rsqrtf(red[0] / (float)D_ + 1e-6f);
  float* ro = res_out + (size_t)row * D_;
  bf16_t* ho = h_out + (size_t)row * D_;
#pragma unroll
  for (int i = 0; i < 8; i++) {
    int idx = tid + i * 256;
    ro[idx] = v[i];
    ho[idx] = (bf16_t)(v[i] * inv * w[idx]);
  }
}

// ---------------- generic bf16 WMMA GEMM (async double-buffered) ----------------
// C[M][N] = A[M][Kd] @ Bt[N][Kd]^T   (A row-major bf16, Bt = B transposed bf16)
// mode 0: outf = acc + bias[n] (bias may be null)
// mode 1: outb = bf16( silu(aux[m][n]) * acc )        (SwiGLU up-proj fuse)
// mode 2: outf = mask[m] ? acc : aux[m][n]            (gen/und select fuse)
#define BKG  64
#define LDSS 72   // 64 + 8 pad (144B row stride, 16B aligned)

__global__ __launch_bounds__(256) void gemm_bf16_kernel(
    const bf16_t* __restrict__ A, const bf16_t* __restrict__ Bt,
    int M, int N, int Kd,
    const float* __restrict__ bias, const float* __restrict__ aux,
    const unsigned char* __restrict__ mask,
    float* __restrict__ outf, bf16_t* __restrict__ outb, int mode) {
  __shared__ bf16_t As[2][128 * LDSS];
  __shared__ bf16_t Bs[2][128 * LDSS];
  const int tid  = threadIdx.x;
  const int lane = tid & 31;
  const int wave = tid >> 5;
  const int wm   = wave >> 2;       // 0..1 -> 64 rows each
  const int wn   = wave & 3;        // 0..3 -> 32 cols each
  const int m0   = blockIdx.y * 128;
  const int n0   = blockIdx.x * 128;
  const int half = lane >> 4;
  const int nl   = lane & 15;
  const int ao   = half ? 16 : 0;   // A-frag K offset
  const int bo   = half ? 16 : 0;   // B-frag K offset

  // per-thread copy descriptors: tile = 128 rows x 64 bf16 = 1024 segs of 16B,
  // 4 segs per thread; 8 consecutive tids cover one 128B row.
  int crow[4], ccol[4];
  unsigned ldsA[2][4], ldsB[2][4];
  const bf16_t* gA[4];
  const bf16_t* gB[4];
#pragma unroll
  for (int i = 0; i < 4; i++) {
    int s = i * 256 + tid;
    crow[i] = s >> 3;
    ccol[i] = (s & 7) * 8;
    gA[i] = A  + (size_t)(m0 + crow[i]) * Kd + ccol[i];
    gB[i] = Bt + (size_t)(n0 + crow[i]) * Kd + ccol[i];
#pragma unroll
    for (int bi = 0; bi < 2; bi++) {
      ldsA[bi][i] = (unsigned)(size_t)&As[bi][crow[i] * LDSS + ccol[i]];
      ldsB[bi][i] = (unsigned)(size_t)&Bs[bi][crow[i] * LDSS + ccol[i]];
    }
  }

  v8f acc[4][2];
#pragma unroll
  for (int i = 0; i < 4; i++)
#pragma unroll
    for (int j = 0; j < 2; j++) acc[i][j] = vzero();

  const int steps = Kd / BKG;

  // prologue: stage tile 0 into buffer 0
#pragma unroll
  for (int i = 0; i < 4; i++) {
    async_copy_b128(ldsA[0][i], gA[i]);
    async_copy_b128(ldsB[0][i], gB[i]);
  }
  wait_async0();
  __syncthreads();

  for (int ks = 0; ks < steps; ++ks) {
    const int cur = ks & 1;
    // kick off next tile into the other buffer (overlaps with compute below)
    if (ks + 1 < steps) {
      const int k0n = (ks + 1) * BKG;
#pragma unroll
      for (int i = 0; i < 4; i++) {
        async_copy_b128(ldsA[1 - cur][i], gA[i] + k0n);
        async_copy_b128(ldsB[1 - cur][i], gB[i] + k0n);
      }
    }
    const bf16_t* as = As[cur];
    const bf16_t* bs = Bs[cur];
#pragma unroll
    for (int kk = 0; kk < 2; kk++) {
      FragA af[4], bfr[2];
#pragma unroll
      for (int fm = 0; fm < 4; fm++)
        af[fm] = load_a(&as[(wm * 64 + fm * 16 + nl) * LDSS + kk * 32 + ao]);
#pragma unroll
      for (int fn = 0; fn < 2; fn++)
        bfr[fn] = load_b(&bs[(wn * 32 + fn * 16 + nl) * LDSS + kk * 32 + bo]);
#pragma unroll
      for (int fm = 0; fm < 4; fm++)
#pragma unroll
        for (int fn = 0; fn < 2; fn++)
          acc[fm][fn] = wmma_bf16(af[fm], bfr[fn], acc[fm][fn]);
    }
    if (ks + 1 < steps) {
      wait_async0();        // our async writes into next buffer are done
      __syncthreads();      // everyone's writes visible, reads of cur done
    }
  }

  // epilogue: lane holds (row = fm*16 + j + 8*half, col = fn*16 + nl)
#pragma unroll
  for (int fm = 0; fm < 4; fm++) {
#pragma unroll
    for (int fn = 0; fn < 2; fn++) {
      const int n = n0 + wn * 32 + fn * 16 + nl;
#pragma unroll
      for (int j = 0; j < 8; j++) {
        const int m = m0 + wm * 64 + fm * 16 + j + 8 * half;
        const float v = acc[fm][fn][j];
        const size_t idx = (size_t)m * N + n;
        if (mode == 0) {
          outf[idx] = v + (bias ? bias[n] : 0.0f);
        } else if (mode == 1) {
          const float g = aux[idx];
          const float s = g / (1.0f + expf(-g));   // silu(gate)
          outb[idx] = (bf16_t)(s * v);
        } else {
          outf[idx] = mask[m] ? v : aux[idx];
        }
      }
    }
  }
}

// ---------------- RoPE + QKV unpack ----------------
__global__ __launch_bounds__(128) void rope_unpack_kernel(
    const float* __restrict__ qkv, const int* __restrict__ positions,
    bf16_t* __restrict__ qo, bf16_t* __restrict__ ko, bf16_t* __restrict__ vTo) {
  const int t = blockIdx.x;        // 0..M-1
  const int b = t >> 11;
  const int l = t & (L_ - 1);
  const float pos = (float)positions[t];
  const float* row = qkv + (size_t)t * 3072;
  const float LOG2_THETA = 19.931568569324174f;   // log2(1e6)

#pragma unroll
  for (int it = 0; it < 8; ++it) {               // Q: 16 heads x 64 pairs
    int pidx = threadIdx.x + it * 128;
    int hh = pidx >> 6;
    int i = pidx & 63;
    float inv_freq = exp2f(-(float)i * (LOG2_THETA / 64.0f));
    float fr = pos * inv_freq, sn, cs;
    sincosf(fr, &sn, &cs);
    float x1 = row[hh * 128 + i];
    float x2 = row[hh * 128 + 64 + i];
    size_t base = ((size_t)(b * H_ + hh) * L_ + l) * HD_;
    qo[base + i]      = (bf16_t)(x1 * cs - x2 * sn);
    qo[base + 64 + i] = (bf16_t)(x2 * cs + x1 * sn);
  }
#pragma unroll
  for (int it = 0; it < 2; ++it) {               // K: 4 heads x 64 pairs
    int pidx = threadIdx.x + it * 128;
    int hh = pidx >> 6;
    int i = pidx & 63;
    float inv_freq = exp2f(-(float)i * (LOG2_THETA / 64.0f));
    float fr = pos * inv_freq, sn, cs;
    sincosf(fr, &sn, &cs);
    float x1 = row[2048 + hh * 128 + i];
    float x2 = row[2048 + hh * 128 + 64 + i];
    size_t base = ((size_t)(b * KV_ + hh) * L_ + l) * HD_;
    ko[base + i]      = (bf16_t)(x1 * cs - x2 * sn);
    ko[base + 64 + i] = (bf16_t)(x2 * cs + x1 * sn);
  }
#pragma unroll
  for (int it = 0; it < 4; ++it) {               // V transpose store
    int e = threadIdx.x + it * 128;              // 0..511
    int kvh = e >> 7;
    int d = e & 127;
    vTo[((size_t)(b * KV_ + kvh) * HD_ + d) * L_ + l] =
        (bf16_t)row[2560 + kvh * 128 + d];
  }
}

// ---------------- flash attention (causal, GQA) ----------------
__global__ __launch_bounds__(128) void attn_kernel(
    const bf16_t* __restrict__ Q, const bf16_t* __restrict__ Kc,
    const bf16_t* __restrict__ Vt, bf16_t* __restrict__ Ob) {
  __shared__ bf16_t Ps[4][16][72];   // per-wave P staging (C-layout -> A-layout)
  const int qt = blockIdx.x;
  const int h  = blockIdx.y;
  const int b  = blockIdx.z;
  const int kv = h >> 2;             // H/KV = 4
  const int tid  = threadIdx.x;
  const int lane = tid & 31;
  const int wave = tid >> 5;
  const int half = lane >> 4;
  const int nl   = lane & 15;
  const int ao   = half ? 16 : 0;
  const int bo   = half ? 16 : 0;
  const int qbase = qt * 64 + wave * 16;
  const float SL2E = 0.08838834764831845f * 1.4426950408889634f; // 1/sqrt(128)*log2e

  FragA qf[4];
  const bf16_t* qp = Q + ((size_t)(b * H_ + h) * L_ + qbase + nl) * HD_;
#pragma unroll
  for (int c = 0; c < 4; c++) qf[c] = load_a(qp + c * 32 + ao);

  v8f oacc[8];
#pragma unroll
  for (int i = 0; i < 8; i++) oacc[i] = vzero();
  float mrow[8], lrow[8];
#pragma unroll
  for (int j = 0; j < 8; j++) { mrow[j] = -INFINITY; lrow[j] = 0.f; }

  const bf16_t* kb0 = Kc + (size_t)(b * KV_ + kv) * L_ * HD_;
  const bf16_t* vb0 = Vt + (size_t)(b * KV_ + kv) * HD_ * L_;

  for (int kt = 0; kt <= qt; ++kt) {
    const int kstart = kt * 64;
    v8f s[4];
#pragma unroll
    for (int i = 0; i < 4; i++) s[i] = vzero();
#pragma unroll
    for (int fn = 0; fn < 4; fn++) {
      const bf16_t* kr = kb0 + (size_t)(kstart + fn * 16 + nl) * HD_;
#pragma unroll
      for (int c = 0; c < 4; c++) {
        FragA kf = load_b(kr + c * 32 + bo);
        s[fn] = wmma_bf16(qf[c], kf, s[fn]);
      }
    }
    if (kt == qt) {   // causal mask on the diagonal tile
#pragma unroll
      for (int fn = 0; fn < 4; fn++) {
        const int kg = kstart + fn * 16 + nl;
#pragma unroll
        for (int j = 0; j < 8; j++) {
          const int rg = qbase + j + 8 * half;
          if (kg > rg) s[fn][j] = -INFINITY;
        }
      }
    }
    float vmax[8];
#pragma unroll
    for (int j = 0; j < 8; j++)
      vmax[j] = fmaxf(fmaxf(s[0][j], s[1][j]), fmaxf(s[2][j], s[3][j]));
#pragma unroll
    for (int off = 1; off < 16; off <<= 1)
#pragma unroll
      for (int j = 0; j < 8; j++)
        vmax[j] = fmaxf(vmax[j], __shfl_xor(vmax[j], off, 32));
    float corr[8];
#pragma unroll
    for (int j = 0; j < 8; j++) {
      float mn = fmaxf(mrow[j], vmax[j]);
      corr[j] = exp2f((mrow[j] - mn) * SL2E);
      mrow[j] = mn;
    }
    float ladd[8];
#pragma unroll
    for (int j = 0; j < 8; j++) ladd[j] = 0.f;
#pragma unroll
    for (int fn = 0; fn < 4; fn++)
#pragma unroll
      for (int j = 0; j < 8; j++) {
        float p = exp2f((s[fn][j] - mrow[j]) * SL2E);
        s[fn][j] = p;
        ladd[j] += p;
      }
#pragma unroll
    for (int off = 1; off < 16; off <<= 1)
#pragma unroll
      for (int j = 0; j < 8; j++) ladd[j] += __shfl_xor(ladd[j], off, 32);
#pragma unroll
    for (int j = 0; j < 8; j++) lrow[j] = lrow[j] * corr[j] + ladd[j];
#pragma unroll
    for (int fo = 0; fo < 8; fo++)
#pragma unroll
      for (int j = 0; j < 8; j++) oacc[fo][j] *= corr[j];
    // stage P through LDS to convert C-layout f32 -> A-layout bf16
#pragma unroll
    for (int fn = 0; fn < 4; fn++)
#pragma unroll
      for (int j = 0; j < 8; j++)
        Ps[wave][j + 8 * half][fn * 16 + nl] = (bf16_t)s[fn][j];
    __syncthreads();
#pragma unroll
    for (int c = 0; c < 2; c++) {
      FragA pf = load_a(&Ps[wave][nl][c * 32 + ao]);
#pragma unroll
      for (int fo = 0; fo < 8; fo++) {
        FragA vf = load_b(vb0 + (size_t)(fo * 16 + nl) * L_ + kstart + c * 32 + bo);
        oacc[fo] = wmma_bf16(pf, vf, oacc[fo]);
      }
    }
    __syncthreads();
  }
  float inv[8];
#pragma unroll
  for (int j = 0; j < 8; j++) inv[j] = 1.0f / lrow[j];
#pragma unroll
  for (int fo = 0; fo < 8; fo++)
#pragma unroll
    for (int j = 0; j < 8; j++) {
      const size_t tok = (size_t)b * L_ + qbase + j + 8 * half;
      Ob[tok * (H_ * HD_) + h * HD_ + fo * 16 + nl] = (bf16_t)(oacc[fo][j] * inv[j]);
    }
}

// ---------------- host-side orchestration ----------------
extern "C" void kernel_launch(void* const* d_in, const int* in_sizes, int n_in,
                              void* d_out, int out_size, void* d_ws, size_t ws_size,
                              hipStream_t stream) {
  (void)in_sizes; (void)n_in; (void)out_size; (void)ws_size;
  const int*   positions = (const int*)  d_in[0];
  const float* hidden    = (const float*)d_in[1];
  const float* resid_in  = (const float*)d_in[2];
  const unsigned char* gmask = (const unsigned char*)d_in[3];
  const float* w_in_ln   = (const float*)d_in[4];
  const float* w_post_ln = (const float*)d_in[5];
  const float* wqkv      = (const float*)d_in[6];
  const float* bqkv      = (const float*)d_in[7];
  const float* wo        = (const float*)d_in[8];
  const float* mlp_gate  = (const float*)d_in[9];
  const float* mlp_up    = (const float*)d_in[10];
  const float* mlp_down  = (const float*)d_in[11];
  const float* gen_gate  = (const float*)d_in[12];
  const float* gen_up    = (const float*)d_in[13];
  const float* gen_down  = (const float*)d_in[14];

  float* out_f = (float*)d_out;                       // (out, residual) concat
  float* res_f = out_f + (size_t)M_ * D_;

  char* ws = (char*)d_ws;
  size_t off = 0;
  auto alloc = [&](size_t bytes) -> void* {
    void* p = ws + off;
    off += (bytes + 255) & ~(size_t)255;
    return p;
  };
  float*  resid1 = (float*) alloc((size_t)M_ * D_ * 4);
  bf16_t* hbf    = (bf16_t*)alloc((size_t)M_ * D_ * 2);
  bf16_t* h2bf   = (bf16_t*)alloc((size_t)M_ * D_ * 2);
  bf16_t* wqkvT  = (bf16_t*)alloc((size_t)3072 * D_ * 2);
  bf16_t* woT    = (bf16_t*)alloc((size_t)D_ * D_ * 2);
  bf16_t* wgT    = (bf16_t*)alloc((size_t)I_ * D_ * 2);   // reused per branch
  bf16_t* wuT    = (bf16_t*)alloc((size_t)I_ * D_ * 2);
  bf16_t* wdT    = (bf16_t*)alloc((size_t)D_ * I_ * 2);
  float*  bigtmp = (float*) alloc((size_t)M_ * I_ * 4);   // qkv, then gate acts
  bf16_t* qbfp   = (bf16_t*)alloc((size_t)B_ * H_ * L_ * HD_ * 2);
  bf16_t* kbfp   = (bf16_t*)alloc((size_t)B_ * KV_ * L_ * HD_ * 2);
  bf16_t* vTbf   = (bf16_t*)alloc((size_t)B_ * KV_ * HD_ * L_ * 2);
  bf16_t* aobf   = (bf16_t*)alloc((size_t)M_ * D_ * 2);
  float*  oproj  = (float*) alloc((size_t)M_ * D_ * 4);
  bf16_t* actb   = (bf16_t*)alloc((size_t)M_ * I_ * 2);
  float*  undb   = (float*) alloc((size_t)M_ * D_ * 4);

  const dim3 tb(32, 8);

  // attention-path weight conversion
  transpose_to_bf16_kernel<<<dim3(3072 / 32, D_ / 32), tb, 0, stream>>>(wqkv, wqkvT, D_, 3072);
  transpose_to_bf16_kernel<<<dim3(D_ / 32,  D_ / 32), tb, 0, stream>>>(wo,   woT,   D_, D_);

  // residual add + input RMSNorm
  add_rmsnorm_kernel<<<M_, 256, 0, stream>>>(hidden, resid_in, w_in_ln, resid1, hbf);

  // QKV projection (+bias)
  gemm_bf16_kernel<<<dim3(3072 / 128, M_ / 128), 256, 0, stream>>>(
      hbf, wqkvT, M_, 3072, D_, bqkv, nullptr, nullptr, bigtmp, nullptr, 0);

  // RoPE + head unpack (+ V transpose)
  rope_unpack_kernel<<<M_, 128, 0, stream>>>(bigtmp, positions, qbfp, kbfp, vTbf);

  // causal flash attention
  attn_kernel<<<dim3(L_ / 64, H_, B_), 128, 0, stream>>>(qbfp, kbfp, vTbf, aobf);

  // output projection
  gemm_bf16_kernel<<<dim3(D_ / 128, M_ / 128), 256, 0, stream>>>(
      aobf, woT, M_, D_, D_, nullptr, nullptr, nullptr, oproj, nullptr, 0);

  // second residual + post RMSNorm (residual goes straight to d_out)
  add_rmsnorm_kernel<<<M_, 256, 0, stream>>>(oproj, resid1, w_post_ln, res_f, h2bf);

  // ---- understanding (und) MLP branch ----
  transpose_to_bf16_kernel<<<dim3(I_ / 32, D_ / 32), tb, 0, stream>>>(mlp_gate, wgT, D_, I_);
  transpose_to_bf16_kernel<<<dim3(I_ / 32, D_ / 32), tb, 0, stream>>>(mlp_up,   wuT, D_, I_);
  transpose_to_bf16_kernel<<<dim3(D_ / 32, I_ / 32), tb, 0, stream>>>(mlp_down, wdT, I_, D_);
  gemm_bf16_kernel<<<dim3(I_ / 128, M_ / 128), 256, 0, stream>>>(
      h2bf, wgT, M_, I_, D_, nullptr, nullptr, nullptr, bigtmp, nullptr, 0);
  gemm_bf16_kernel<<<dim3(I_ / 128, M_ / 128), 256, 0, stream>>>(
      h2bf, wuT, M_, I_, D_, nullptr, bigtmp, nullptr, nullptr, actb, 1);
  gemm_bf16_kernel<<<dim3(D_ / 128, M_ / 128), 256, 0, stream>>>(
      actb, wdT, M_, D_, I_, nullptr, nullptr, nullptr, undb, nullptr, 0);

  // ---- generation (gen) MLP branch; final select fused into down GEMM ----
  transpose_to_bf16_kernel<<<dim3(I_ / 32, D_ / 32), tb, 0, stream>>>(gen_gate, wgT, D_, I_);
  transpose_to_bf16_kernel<<<dim3(I_ / 32, D_ / 32), tb, 0, stream>>>(gen_up,   wuT, D_, I_);
  transpose_to_bf16_kernel<<<dim3(D_ / 32, I_ / 32), tb, 0, stream>>>(gen_down, wdT, I_, D_);
  gemm_bf16_kernel<<<dim3(I_ / 128, M_ / 128), 256, 0, stream>>>(
      h2bf, wgT, M_, I_, D_, nullptr, nullptr, nullptr, bigtmp, nullptr, 0);
  gemm_bf16_kernel<<<dim3(I_ / 128, M_ / 128), 256, 0, stream>>>(
      h2bf, wuT, M_, I_, D_, nullptr, bigtmp, nullptr, nullptr, actb, 1);
  gemm_bf16_kernel<<<dim3(D_ / 128, M_ / 128), 256, 0, stream>>>(
      actb, wdT, M_, D_, I_, nullptr, undb, gmask, out_f, nullptr, 2);
}
